// MHABlock_3539053052293
// MI455X (gfx1250) — matrix operand order repaired and checked by reading, not compiled, not run on hardware
//
#include <hip/hip_runtime.h>
#include <hip/hip_bf16.h>
#include <math.h>

typedef __attribute__((ext_vector_type(16))) _Float16 v16h;
typedef __attribute__((ext_vector_type(8)))  _Float16 v8h;
typedef __attribute__((ext_vector_type(8)))  float    v8f;

#define DEV __device__ __forceinline__

// ---------------- WMMA helpers (CDNA5 16x16x32 f16 -> f32) ----------------

DEV v8f wmma_f16(v16h a, v16h b, v8f c) {
  return __builtin_amdgcn_wmma_f32_16x16x32_f16(
      /*neg_a=*/false, a, /*neg_b=*/false, b,
      /*c_mod=*/(short)0, c, /*reuse_a=*/false, /*reuse_b=*/false);
}

// A fragment: 16x32 (MxK) tile from a row-major f16 matrix, ld in elements.
DEV v16h load_a_frag(const _Float16* base, int ld) {
  int lane = threadIdx.x & 31;
  int m    = lane & 15;
  int ko   = (lane >> 4) << 3;           // 0 or 8
  const _Float16* p = base + (size_t)m * ld + ko;
  v8h lo = *(const v8h*)(p);
  v8h hi = *(const v8h*)(p + 16);
  v16h a;
#pragma unroll
  for (int i = 0; i < 8; ++i) { a[i] = lo[i]; a[i + 8] = hi[i]; }
  return a;
}

// B fragment: 32x16 (KxN) tile sourced from an N-major matrix bt[N][K].
DEV v16h load_b_frag(const _Float16* bt, int ld) {
  int lane = threadIdx.x & 31;
  int n    = lane & 15;
  int kb   = (lane >> 4) << 3;           // 0 or 8
  const _Float16* p = bt + (size_t)n * ld + kb;
  v8h lo = *(const v8h*)(p);
  v8h hi = *(const v8h*)(p + 16);
  v16h b;
#pragma unroll
  for (int i = 0; i < 8; ++i) { b[i] = lo[i]; b[i + 8] = hi[i]; }
  return b;
}

// Issue one 16-byte async global->LDS copy. ldsoff = byte offset from the
// workgroup's LDS base (ISA 15.18: the VDST VGPR plus the instruction offset, on top of the LDS base).
DEV void async_cp16(unsigned ldsoff, const _Float16* gp) {
  asm volatile("global_load_async_to_lds_b128 %0, %1, off"
               :: "v"(ldsoff), "v"(gp) : "memory");
}
DEV void async_wait0() {
  asm volatile("s_wait_asynccnt 0x0" ::: "memory");
}

// ---------------- CDNA5 ISA probe (compiled, never launched) ----------------

__global__ void k_probe_cdna5(const float* __restrict__ g, float* __restrict__ out) {
  __shared__ __align__(16) float stage[256];
  unsigned ldsoff = threadIdx.x * 16u;
  const float* gp = g + threadIdx.x * 4;
  asm volatile("global_load_async_to_lds_b128 %0, %1, off"
               :: "v"(ldsoff), "v"(gp) : "memory");
  asm volatile("s_wait_asynccnt 0x0" ::: "memory");
  asm volatile("tensor_load_to_lds s[8:11], s[12:19]" ::: "memory");
  __builtin_amdgcn_s_wait_tensorcnt(0);
  __syncthreads();
  out[threadIdx.x] = stage[threadIdx.x];
}

// ---------------- Weight transpose+convert: dst[n*K+k] = (f16)src[k*N+n] ----

__global__ void k_transpose_f16(const float* __restrict__ src,
                                _Float16* __restrict__ dst, int K, int N) {
  int total = K * N;
  for (int i = blockIdx.x * blockDim.x + threadIdx.x; i < total;
       i += gridDim.x * blockDim.x) {
    int n = i / K, k = i - n * K;
    dst[(size_t)n * K + k] = (_Float16)src[(size_t)k * N + n];
  }
}

// ---------------- RMS norm ----------------

__global__ void k_rms_to_f16(const float* __restrict__ x,
                             const float* __restrict__ sc,
                             _Float16* __restrict__ out, int D) {
  int row = blockIdx.x;
  const float* xr = x + (size_t)row * D;
  __shared__ float red[8];
  float ss = 0.f;
  for (int d = threadIdx.x; d < D; d += blockDim.x) { float v = xr[d]; ss += v * v; }
#pragma unroll
  for (int m = 16; m >= 1; m >>= 1) ss += __shfl_xor(ss, m, 32);
  if ((threadIdx.x & 31) == 0) red[threadIdx.x >> 5] = ss;
  __syncthreads();
  float tot = 0.f;
#pragma unroll
  for (int i = 0; i < 8; ++i) tot += red[i];
  float r = rsqrtf(tot / (float)D + 1e-6f);
  for (int d = threadIdx.x; d < D; d += blockDim.x)
    out[(size_t)row * D + d] = (_Float16)(xr[d] * r * sc[d]);
}

__global__ void k_rms_f32(const float* __restrict__ x,
                          const float* __restrict__ sc,
                          float* __restrict__ out, int D) {
  int row = blockIdx.x;
  const float* xr = x + (size_t)row * D;
  __shared__ float red[8];
  float ss = 0.f;
  for (int d = threadIdx.x; d < D; d += blockDim.x) { float v = xr[d]; ss += v * v; }
#pragma unroll
  for (int m = 16; m >= 1; m >>= 1) ss += __shfl_xor(ss, m, 32);
  if ((threadIdx.x & 31) == 0) red[threadIdx.x >> 5] = ss;
  __syncthreads();
  float tot = 0.f;
#pragma unroll
  for (int i = 0; i < 8; ++i) tot += red[i];
  float r = rsqrtf(tot / (float)D + 1e-6f);
  for (int d = threadIdx.x; d < D; d += blockDim.x)
    out[(size_t)row * D + d] = xr[d] * r * sc[d];
}

// ---------------- Pipelined WMMA GEMM: C = A(f16,MxK) * Bt(f16,NxK) ----------
// one wave -> 16x64 tile; ping-pong register buffers (no rotation moves).

__global__ void __launch_bounds__(256, 1)
k_gemm(const _Float16* __restrict__ A, const _Float16* __restrict__ Bt,
       float* __restrict__ C, const float* __restrict__ bias,
       int M, int N, int K) {
  int warp = threadIdx.x >> 5;
  int ntiles = N >> 6;
  int wid = blockIdx.x * 8 + warp;
  int mt = wid / ntiles, nt = wid - mt * ntiles;
  int m0 = mt << 4, n0 = nt << 6;
  if (m0 >= M) return;

  v8f acc[4];
#pragma unroll
  for (int j = 0; j < 4; ++j)
#pragma unroll
    for (int r = 0; r < 8; ++r) acc[j][r] = 0.f;

  const _Float16* ap  = A + (size_t)m0 * K;
  const _Float16* bp0 = Bt + (size_t)n0 * K;

  v16h a0 = load_a_frag(ap, K);
  v16h b0[4];
#pragma unroll
  for (int j = 0; j < 4; ++j) b0[j] = load_b_frag(bp0 + (size_t)(j * 16) * K, K);

  // K assumed a multiple of 64 (1536 here): 2x-unrolled ping-pong.
  for (int k0 = 0; k0 < K; k0 += 64) {
    v16h a1 = load_a_frag(ap + k0 + 32, K);
    v16h b1[4];
#pragma unroll
    for (int j = 0; j < 4; ++j)
      b1[j] = load_b_frag(bp0 + (size_t)(j * 16) * K + k0 + 32, K);
#pragma unroll
    for (int j = 0; j < 4; ++j) acc[j] = wmma_f16(a0, b0[j], acc[j]);

    if (k0 + 64 < K) {
      a0 = load_a_frag(ap + k0 + 64, K);
#pragma unroll
      for (int j = 0; j < 4; ++j)
        b0[j] = load_b_frag(bp0 + (size_t)(j * 16) * K + k0 + 64, K);
    }
#pragma unroll
    for (int j = 0; j < 4; ++j) acc[j] = wmma_f16(a1, b1[j], acc[j]);
  }

  int lane = threadIdx.x & 31;
  int n = lane & 15, hi = lane >> 4;
#pragma unroll
  for (int j = 0; j < 4; ++j) {
    float bv = bias ? bias[n0 + j * 16 + n] : 0.f;
#pragma unroll
    for (int r = 0; r < 8; ++r) {
      int mm = m0 + r + 8 * hi;
      C[(size_t)mm * N + n0 + j * 16 + n] = acc[j][r] + bv;
    }
  }
}

// ---------------- QKV epilogue: layernorm + RoPE ----------------

DEV void rope2(float* v, int c0, int s) {
  const float lnb = 0.14290784f;          // ln(8129)/63
#pragma unroll
  for (int pp = 0; pp < 2; ++pp) {
    int f = (c0 >> 1) + pp;
    float base = __expf(lnb * (float)f);
    float invf = 1.f / ((float)f + base);
    float th = (float)s * invf;
    float sn, cs;
    __sincosf(th, &sn, &cs);
    float e = v[2 * pp], od = v[2 * pp + 1];
    v[2 * pp]     = e * cs - od * sn;
    v[2 * pp + 1] = e * sn + od * cs;
  }
}

__global__ void k_qkv_epilogue(const float* __restrict__ qkv,
                               const float* __restrict__ lnqs, const float* __restrict__ lnqb,
                               const float* __restrict__ lnks, const float* __restrict__ lnkb,
                               const float* __restrict__ lnvs, const float* __restrict__ lnvb,
                               _Float16* __restrict__ q16, _Float16* __restrict__ k16,
                               _Float16* __restrict__ v16t, int S) {
  int s = blockIdx.x;
  int warp = threadIdx.x >> 5, lane = threadIdx.x & 31;
  const float* row = qkv + (size_t)s * 1344;

  {   // q: wave w -> head w
    const float* src = row + warp * 128;
    int c0 = lane * 4;
    float v[4];
#pragma unroll
    for (int j = 0; j < 4; ++j) v[j] = src[c0 + j];
    float sm = 0.f, sq = 0.f;
#pragma unroll
    for (int j = 0; j < 4; ++j) { sm += v[j]; sq += v[j] * v[j]; }
#pragma unroll
    for (int m = 16; m >= 1; m >>= 1) { sm += __shfl_xor(sm, m, 32); sq += __shfl_xor(sq, m, 32); }
    float mean = sm * (1.f / 128.f);
    float var  = sq * (1.f / 128.f) - mean * mean;
    float rs   = rsqrtf(var + 1e-5f);
#pragma unroll
    for (int j = 0; j < 4; ++j) v[j] = (v[j] - mean) * rs * lnqs[c0 + j] + lnqb[c0 + j];
    rope2(v, c0, s);
    _Float16* dst = q16 + (size_t)s * 1024 + warp * 128 + c0;
#pragma unroll
    for (int j = 0; j < 4; ++j) dst[j] = (_Float16)v[j];
  }

  if (warp == 0) {   // k
    const float* src = row + 1024;
    int c0 = lane * 4;
    float v[4];
#pragma unroll
    for (int j = 0; j < 4; ++j) v[j] = src[c0 + j];
    float sm = 0.f, sq = 0.f;
#pragma unroll
    for (int j = 0; j < 4; ++j) { sm += v[j]; sq += v[j] * v[j]; }
#pragma unroll
    for (int m = 16; m >= 1; m >>= 1) { sm += __shfl_xor(sm, m, 32); sq += __shfl_xor(sq, m, 32); }
    float mean = sm * (1.f / 128.f);
    float var  = sq * (1.f / 128.f) - mean * mean;
    float rs   = rsqrtf(var + 1e-5f);
#pragma unroll
    for (int j = 0; j < 4; ++j) v[j] = (v[j] - mean) * rs * lnks[c0 + j] + lnkb[c0 + j];
    rope2(v, c0, s);
    _Float16* dst = k16 + (size_t)s * 128 + c0;
#pragma unroll
    for (int j = 0; j < 4; ++j) dst[j] = (_Float16)v[j];
  }

  if (warp == 1) {   // v -> transposed (DV x S)
    const float* src = row + 1152;
    int c0 = lane * 6;
    float v[6];
#pragma unroll
    for (int j = 0; j < 6; ++j) v[j] = src[c0 + j];
    float sm = 0.f, sq = 0.f;
#pragma unroll
    for (int j = 0; j < 6; ++j) { sm += v[j]; sq += v[j] * v[j]; }
#pragma unroll
    for (int m = 16; m >= 1; m >>= 1) { sm += __shfl_xor(sm, m, 32); sq += __shfl_xor(sq, m, 32); }
    float mean = sm * (1.f / 192.f);
    float var  = sq * (1.f / 192.f) - mean * mean;
    float rs   = rsqrtf(var + 1e-5f);
#pragma unroll
    for (int j = 0; j < 6; ++j) {
      float val = (v[j] - mean) * rs * lnvs[c0 + j] + lnvb[c0 + j];
      v16t[(size_t)(c0 + j) * S + s] = (_Float16)val;
    }
  }
}

// ---------------- Flash attention, block-cooperative -----------------------
// Block = 8 waves = 128 q-rows of one head.  K/V tiles (32 kv cols) staged in
// double-buffered dynamic LDS via async global->LDS copies; bias streamed
// from HBM with prefetch (the 536MB bias stream is the roofline floor).

#define KT_BYTES (32 * 128 * 2)            // 8 KB  K tile  [kv][dk]
#define VT_BYTES (192 * 32 * 2)            // 12 KB V tile  [dv][kv]
#define BUF_BYTES (KT_BYTES + VT_BYTES)    // 20 KB per buffer

// Stage K (32x128) and V (192x32) tiles for kv0 into LDS at byte offset obase.
DEV void stage_kv_async(const _Float16* k16, const _Float16* v16t, int S,
                        int kv0, unsigned obase) {
  int tid = threadIdx.x;
  // K tile: contiguous 8KB -> 512 x 16B chunks, 2 per thread
#pragma unroll
  for (int i = 0; i < 2; ++i) {
    int c = tid * 2 + i;
    async_cp16(obase + c * 16, k16 + (size_t)kv0 * 128 + c * 8);
  }
  // V tile: 192 rows x 64B -> 768 x 16B chunks, 3 per thread
#pragma unroll
  for (int i = 0; i < 3; ++i) {
    int c = tid + i * 256;
    int dv = c >> 2, c4 = c & 3;
    async_cp16(obase + KT_BYTES + c * 16, v16t + (size_t)dv * S + kv0 + c4 * 8);
  }
}

__global__ void __launch_bounds__(256, 1)
k_attention(const _Float16* __restrict__ q16, const _Float16* __restrict__ k16,
            const _Float16* __restrict__ v16t, const float* __restrict__ bias,
            _Float16* __restrict__ y16, int S) {
  extern __shared__ __align__(16) char dynsmem[];
  __shared__ __align__(16) _Float16 plds[8][16][32];

  unsigned dynbase = __builtin_amdgcn_groupstaticsize();

  int warp = threadIdx.x >> 5, lane = threadIdx.x & 31;
  int head = blockIdx.x >> 5;              // 32 q-blocks per head
  int q0   = ((blockIdx.x & 31) << 7) + warp * 16;
  int n = lane & 15, hi = lane >> 4;

  v16h qa[4];
#pragma unroll
  for (int kc = 0; kc < 4; ++kc)
    qa[kc] = load_a_frag(q16 + (size_t)q0 * 1024 + head * 128 + kc * 32, 1024);

  v8f o[12];
#pragma unroll
  for (int j = 0; j < 12; ++j)
#pragma unroll
    for (int r = 0; r < 8; ++r) o[j][r] = 0.f;
  float mrow[8], lrow[8];
#pragma unroll
  for (int r = 0; r < 8; ++r) { mrow[r] = -1e30f; lrow[r] = 0.f; }

  const float scale = 0.08838834764831845f;   // 1/sqrt(128)
  const float* bp = bias + ((size_t)head * S + q0) * S;

  // prologue: stage first tile into buffer 0
  stage_kv_async(k16, v16t, S, 0, dynbase);
  async_wait0();
  __syncthreads();
  int cur = 0;

  for (int kv0 = 0; kv0 < S; kv0 += 32) {
    // kick off next tile into the other buffer
    if (kv0 + 32 < S)
      stage_kv_async(k16, v16t, S, kv0 + 32,
                     dynbase + (cur ? 0u : (unsigned)BUF_BYTES));

    // current buffer pointers (computed as scalars -> no static initializer)
    char* cb = dynsmem + (cur ? BUF_BYTES : 0);
    const _Float16* kt = (const _Float16*)cb;
    const _Float16* vt = (const _Float16*)(cb + KT_BYTES);

    // ---- S = Q K^T (8 WMMAs, operands from LDS) ----
    v8f s0, s1;
#pragma unroll
    for (int r = 0; r < 8; ++r) { s0[r] = 0.f; s1[r] = 0.f; }
#pragma unroll
    for (int kc = 0; kc < 4; ++kc) {
      v16h b0 = load_b_frag(kt + kc * 32, 128);
      v16h b1 = load_b_frag(kt + 16 * 128 + kc * 32, 128);
      s0 = wmma_f16(qa[kc], b0, s0);
      s1 = wmma_f16(qa[kc], b1, s1);
    }

    // ---- prefetch streaming bias two kv-steps ahead ----
    if (kv0 + 64 < S) {
#pragma unroll
      for (int r = 0; r < 8; ++r) {
        int mm = r + 8 * hi;
        __builtin_prefetch(&bp[(size_t)mm * S + kv0 + 64 + n], 0, 0);
        __builtin_prefetch(&bp[(size_t)mm * S + kv0 + 80 + n], 0, 0);
      }
    }

    // ---- scale + bias + tanh soft-cap (5.0) ----
#pragma unroll
    for (int r = 0; r < 8; ++r) {
      int mm = r + 8 * hi;
      float b0v = bp[(size_t)mm * S + kv0 + n];
      float b1v = bp[(size_t)mm * S + kv0 + 16 + n];
      float a0 = s0[r] * scale + b0v;
      float a1 = s1[r] * scale + b1v;
      s0[r] = 5.f * tanhf(a0 * 0.2f);
      s1[r] = 5.f * tanhf(a1 * 0.2f);
    }

    // ---- online softmax ----
    float alpha[8];
#pragma unroll
    for (int r = 0; r < 8; ++r) {
      float rm = fmaxf(s0[r], s1[r]);
#pragma unroll
      for (int m = 8; m >= 1; m >>= 1) rm = fmaxf(rm, __shfl_xor(rm, m, 32));
      float nm = fmaxf(mrow[r], rm);
      alpha[r] = __expf(mrow[r] - nm);
      mrow[r] = nm;
      float p0 = __expf(s0[r] - nm);
      float p1 = __expf(s1[r] - nm);
      s0[r] = p0; s1[r] = p1;
      float rs = p0 + p1;
#pragma unroll
      for (int m = 8; m >= 1; m >>= 1) rs += __shfl_xor(rs, m, 32);
      lrow[r] = lrow[r] * alpha[r] + rs;
    }
#pragma unroll
    for (int j = 0; j < 12; ++j)
#pragma unroll
      for (int r = 0; r < 8; ++r) o[j][r] *= alpha[r];

    // ---- P (C-layout) -> LDS -> A-layout fragment ----
#pragma unroll
    for (int r = 0; r < 8; ++r) {
      int mm = r + 8 * hi;
      plds[warp][mm][n]      = (_Float16)s0[r];
      plds[warp][mm][16 + n] = (_Float16)s1[r];
    }
    asm volatile("s_wait_dscnt 0" ::: "memory");
    v16h pa = load_a_frag(&plds[warp][0][0], 32);

    // ---- O += P V (12 WMMAs, V fragments from LDS) ----
#pragma unroll
    for (int j = 0; j < 12; ++j) {
      v16h vb = load_b_frag(vt + j * 16 * 32, 32);
      o[j] = wmma_f16(pa, vb, o[j]);
    }

    // finish this wave's async copies, publish across the block, swap buffers
    async_wait0();
    __syncthreads();
    cur ^= 1;
  }

  float inv[8];
#pragma unroll
  for (int r = 0; r < 8; ++r) inv[r] = 1.f / lrow[r];
#pragma unroll
  for (int j = 0; j < 12; ++j)
#pragma unroll
    for (int r = 0; r < 8; ++r) {
      int mm = q0 + r + 8 * hi;
      y16[(size_t)mm * 1536 + head * 192 + j * 16 + n] = (_Float16)(o[j][r] * inv[r]);
    }
}

// ---------------- launch ----------------

extern "C" void kernel_launch(void* const* d_in, const int* in_sizes, int n_in,
                              void* d_out, int out_size, void* d_ws, size_t ws_size,
                              hipStream_t stream) {
  const float* x     = (const float*)d_in[0];
  const float* abias = (const float*)d_in[1];
  const float* rmss  = (const float*)d_in[2];
  const float* qw    = (const float*)d_in[3];
  const float* lnqs  = (const float*)d_in[4];
  const float* lnqb  = (const float*)d_in[5];
  const float* kw    = (const float*)d_in[6];
  const float* lnks  = (const float*)d_in[7];
  const float* lnkb  = (const float*)d_in[8];
  const float* vw    = (const float*)d_in[9];
  const float* lnvs  = (const float*)d_in[10];
  const float* lnvb  = (const float*)d_in[11];
  const float* outw  = (const float*)d_in[12];
  const float* outb  = (const float*)d_in[13];
  const float* frms  = (const float*)d_in[14];

  const int S = 4096, D = 1536, H = 8, DK = 128, DV = 192, NQKV = 1344;

  char* w = (char*)d_ws;
  auto alloc = [&](size_t bytes) -> char* {
    char* p = w;
    w += (bytes + 255) & ~(size_t)255;
    return p;
  };
  _Float16* h16  = (_Float16*)alloc((size_t)S * D * 2);
  _Float16* qkvt = (_Float16*)alloc((size_t)NQKV * D * 2);
  _Float16* owt  = (_Float16*)alloc((size_t)D * D * 2);
  float*    qkv  = (float*)   alloc((size_t)S * NQKV * 4);
  _Float16* q16  = (_Float16*)alloc((size_t)S * H * DK * 2);
  _Float16* k16  = (_Float16*)alloc((size_t)S * DK * 2);
  _Float16* v16t = (_Float16*)alloc((size_t)DV * S * 2);
  _Float16* y16  = (_Float16*)alloc((size_t)S * D * 2);
  float*    y2   = (float*)   alloc((size_t)S * D * 4);

  k_transpose_f16<<<256, 256, 0, stream>>>(qw, qkvt, D, H * DK);
  k_transpose_f16<<<64, 256, 0, stream>>>(kw, qkvt + (size_t)1024 * D, D, DK);
  k_transpose_f16<<<64, 256, 0, stream>>>(vw, qkvt + (size_t)1152 * D, D, DV);
  k_transpose_f16<<<256, 256, 0, stream>>>(outw, owt, D, D);

  k_rms_to_f16<<<S, 256, 0, stream>>>(x, rmss, h16, D);

  k_gemm<<<(S / 16) * (NQKV / 64) / 8, 256, 0, stream>>>(h16, qkvt, qkv, nullptr, S, NQKV, D);

  k_qkv_epilogue<<<S, 256, 0, stream>>>(qkv, lnqs, lnqb, lnks, lnkb, lnvs, lnvb,
                                        q16, k16, v16t, S);

  // 8 heads x 32 q-blocks; 2 x 20KB dynamic LDS for double-buffered K/V tiles
  k_attention<<<H * (S / 128), 256, 2 * BUF_BYTES, stream>>>(q16, k16, v16t, abias, y16, S);

  k_gemm<<<(S / 16) * (D / 64) / 8, 256, 0, stream>>>(y16, owt, y2, outb, S, D, D);

  k_rms_f32<<<S, 256, 0, stream>>>(y2, frms, (float*)d_out, D);
}
